// Llama4TextExperts_74457553043908
// MI455X (gfx1250) — matrix operand order repaired and checked by reading, not compiled, not run on hardware
//
#include <hip/hip_runtime.h>

// ---------------------------------------------------------------------------
// Grouped (ragged) MoE FFN with SwiGLU for MI455X (gfx1250, wave32, WMMA).
//   gate = x @ Wg[e]; up = x @ Wu[e]; out = (silu(gate)*up) @ Wd[e]
// bf16 WMMA (v_wmma_f32_16x16x32_bf16) with fp32 accumulate.
//  - x is converted to bf16 once (tiny pre-pass, lives in L2), so the A tile
//    is staged with CDNA5 async global->LDS copies (ASYNCcnt) when available.
//  - Weights are converted fp32->bf16 on the global->LDS path with
//    v_cvt_pk_bf16_f32 (packed) when available.
//  - Intermediate h = silu(gate)*up stored transposed [I][T] bf16 so the WMMA
//    f32 C-layout (8 consecutive M per lane at fixed N) maps to b128 stores.
// ---------------------------------------------------------------------------

#define NUM_EXPERTS 8
#define HIDDEN      1024
#define INTER       2048
#define TOKENS      4096

#define BM   128          // block tile M (tokens)
#define BN   128          // block tile N
#define BK   64           // K chunk staged in LDS
#define PADK 80           // LDS row stride in bf16 elements (BK + 16, keeps 16B align)

typedef float          v8f    __attribute__((ext_vector_type(8)));
typedef __bf16         v16bf  __attribute__((ext_vector_type(16)));
typedef __bf16         bf16x2 __attribute__((ext_vector_type(2)));
typedef unsigned short u16x4  __attribute__((ext_vector_type(4)));
typedef unsigned short u16x8  __attribute__((ext_vector_type(8)));
typedef unsigned short u16x16 __attribute__((ext_vector_type(16)));
typedef unsigned int   u32x2  __attribute__((ext_vector_type(2)));
typedef unsigned int   u32x4  __attribute__((ext_vector_type(4)));
typedef float          f32x4  __attribute__((ext_vector_type(4)));
typedef int            i32x4v __attribute__((vector_size(16)));   // gcc-style v4i32

#if __has_builtin(__builtin_amdgcn_global_load_async_to_lds_b128) && \
    __has_builtin(__builtin_amdgcn_s_wait_asynccnt)
#define HAVE_ASYNC_LDS 1
#else
#define HAVE_ASYNC_LDS 0
#endif

// Pack two fp32 into two bf16 (low = a, high = b).
__device__ __forceinline__ unsigned pk_bf16(float a, float b) {
#if __has_builtin(__builtin_amdgcn_cvt_pk_bf16_f32)
  return __builtin_bit_cast(unsigned, __builtin_amdgcn_cvt_pk_bf16_f32(a, b));
#else
  unsigned ua = __float_as_uint(a) + 0x8000u;   // round-half-up
  unsigned ub = __float_as_uint(b) + 0x8000u;
  // result bytes: {ua.b2, ua.b3, ub.b2, ub.b3}  (sel 0..3 -> s1, 4..7 -> s0)
  return __builtin_amdgcn_perm(ub, ua, 0x07060302u);
#endif
}

// Build a 16x32 (A) or 32x16 (B) bf16 WMMA fragment from two 16B LDS loads.
__device__ __forceinline__ v16bf make_frag(const unsigned short* p0,
                                           const unsigned short* p1) {
  u16x8 lo = *(const u16x8*)p0;
  u16x8 hi = *(const u16x8*)p1;
  u16x16 cat = __builtin_shufflevector(lo, hi, 0, 1, 2, 3, 4, 5, 6, 7,
                                       8, 9, 10, 11, 12, 13, 14, 15);
  return __builtin_bit_cast(v16bf, cat);
}

// Async 16-byte global->LDS copy (ASYNCcnt) with synchronous fallback.
__device__ __forceinline__ void copy16_g2l(const unsigned short* g,
                                           unsigned short* l) {
#if HAVE_ASYNC_LDS
  __builtin_amdgcn_global_load_async_to_lds_b128(
      (__attribute__((address_space(1))) i32x4v*)g,
      (__attribute__((address_space(3))) i32x4v*)l, 0, 0);
#else
  *(u16x8*)l = *(const u16x8*)g;
#endif
}

__device__ __forceinline__ void wait_async_copies() {
#if HAVE_ASYNC_LDS
  __builtin_amdgcn_s_wait_asynccnt(0);
#endif
}

__device__ __forceinline__ int expert_of_row(const int* __restrict__ gsz, int row) {
  int e = NUM_EXPERTS - 1;
  int start = 0;
  #pragma unroll
  for (int i = 0; i < NUM_EXPERTS; ++i) {
    int g = gsz[i];
    if (row < start + g) { e = i; break; }
    start += g;
  }
  return e;
}

// ---------------------------------------------------------------------------
// Kernel 0: xbf[T][H] = bf16(x)   (8 MB, stays hot in the 192 MB L2)
// ---------------------------------------------------------------------------
__global__ __launch_bounds__(256) void cvt_x_kernel(
    const float* __restrict__ x, unsigned short* __restrict__ xbf) {
  size_t idx = ((size_t)blockIdx.x * 256 + threadIdx.x) * 8;
  f32x4 a = *(const f32x4*)(x + idx);
  f32x4 b = *(const f32x4*)(x + idx + 4);
  u32x4 p;
  p[0] = pk_bf16(a[0], a[1]);
  p[1] = pk_bf16(a[2], a[3]);
  p[2] = pk_bf16(b[0], b[1]);
  p[3] = pk_bf16(b[2], b[3]);
  *(u32x4*)(xbf + idx) = p;
}

// ---------------------------------------------------------------------------
// Kernel 1: fused gate+up GEMM + SwiGLU.  Writes hsT[I][T] (bf16 bits).
// ---------------------------------------------------------------------------
__global__ __launch_bounds__(256) void moe_gateup_kernel(
    const unsigned short* __restrict__ xbf, // [T,H] bf16
    const float* __restrict__ Wg,           // [E,H,I]
    const float* __restrict__ Wu,           // [E,H,I]
    const int*   __restrict__ gsz,          // [E]
    unsigned short* __restrict__ hsT)       // [I,T] bf16
{
  __shared__ __align__(16) unsigned short As[BM * PADK];
  __shared__ __align__(16) unsigned short Bg[BN * PADK];
  __shared__ __align__(16) unsigned short Bu[BN * PADK];

  const int m0 = blockIdx.x * BM;
  const int n0 = blockIdx.y * BN;

  const int e = expert_of_row(gsz, m0);   // tiles assumed within one group
  const float* wg = Wg + (size_t)e * HIDDEN * INTER;
  const float* wu = Wu + (size_t)e * HIDDEN * INTER;

  const int tid  = threadIdx.x;
  const int lane = tid & 31;
  const int wid  = tid >> 5;
  const int r    = lane & 15;
  const int hh   = lane >> 4;          // half-wave select
  const int wm   = (wid & 3) * 32;     // wave M offset in block tile
  const int wn   = (wid >> 2) * 64;    // wave N offset in block tile

  v8f accg[2][4];
  v8f accu[2][4];
  const v8f vzero = {0.f, 0.f, 0.f, 0.f, 0.f, 0.f, 0.f, 0.f};
  #pragma unroll
  for (int mt = 0; mt < 2; ++mt)
    #pragma unroll
    for (int nt = 0; nt < 4; ++nt) { accg[mt][nt] = vzero; accu[mt][nt] = vzero; }

  for (int kc = 0; kc < HIDDEN; kc += BK) {
    // --- stage A tile: 128x64 bf16, straight copy via async global->LDS ---
    #pragma unroll
    for (int p = 0; p < 4; ++p) {
      int idx = p * 256 + tid;
      int row = idx >> 3;              // 8 x 16B per row (64 bf16)
      int c8  = (idx & 7) << 3;
      copy16_g2l(xbf + (size_t)(m0 + row) * HIDDEN + kc + c8,
                 &As[row * PADK + c8]);
    }
    // --- stage B tiles: 64x128 fp32 -> bf16 (packed cvt), transpose [n][k] ---
    #pragma unroll
    for (int p = 0; p < 8; ++p) {
      int idx = p * 256 + tid;
      int k  = idx >> 5;               // 32 float4 per row
      int c4 = (idx & 31) << 2;
      size_t go = (size_t)(kc + k) * INTER + n0 + c4;
      f32x4 g = *(const f32x4*)(wg + go);
      f32x4 u = *(const f32x4*)(wu + go);
      if (kc + BK < HIDDEN) {          // prefetch next K chunk of weights
        __builtin_prefetch(wg + go + (size_t)BK * INTER, 0, 0);
        __builtin_prefetch(wu + go + (size_t)BK * INTER, 0, 0);
      }
      unsigned g01 = pk_bf16(g[0], g[1]);
      unsigned g23 = pk_bf16(g[2], g[3]);
      unsigned u01 = pk_bf16(u[0], u[1]);
      unsigned u23 = pk_bf16(u[2], u[3]);
      Bg[(c4 + 0) * PADK + k] = (unsigned short)g01;
      Bg[(c4 + 1) * PADK + k] = (unsigned short)(g01 >> 16);
      Bg[(c4 + 2) * PADK + k] = (unsigned short)g23;
      Bg[(c4 + 3) * PADK + k] = (unsigned short)(g23 >> 16);
      Bu[(c4 + 0) * PADK + k] = (unsigned short)u01;
      Bu[(c4 + 1) * PADK + k] = (unsigned short)(u01 >> 16);
      Bu[(c4 + 2) * PADK + k] = (unsigned short)u23;
      Bu[(c4 + 3) * PADK + k] = (unsigned short)(u23 >> 16);
    }
    wait_async_copies();
    __syncthreads();

    #pragma unroll
    for (int ks = 0; ks < BK; ks += 32) {
      v16bf afrag[2];
      #pragma unroll
      for (int mt = 0; mt < 2; ++mt) {
        int base = (wm + mt * 16 + r) * PADK + ks + 8 * hh;
        afrag[mt] = make_frag(&As[base], &As[base + 16]);
      }
      #pragma unroll
      for (int nt = 0; nt < 4; ++nt) {
        int base = (wn + nt * 16 + r) * PADK + ks + 8 * hh;
        v16bf bg = make_frag(&Bg[base], &Bg[base + 16]);
        v16bf bu = make_frag(&Bu[base], &Bu[base + 16]);
        #pragma unroll
        for (int mt = 0; mt < 2; ++mt) {
          accg[mt][nt] = __builtin_amdgcn_wmma_f32_16x16x32_bf16(
              false, afrag[mt], false, bg, (short)0, accg[mt][nt], false, false);
          accu[mt][nt] = __builtin_amdgcn_wmma_f32_16x16x32_bf16(
              false, afrag[mt], false, bu, (short)0, accu[mt][nt], false, false);
        }
      }
    }
    __syncthreads();
  }

  // --- epilogue: h = silu(gate) * up, store transposed hsT[n][m] as bf16 ---
  #pragma unroll
  for (int mt = 0; mt < 2; ++mt) {
    #pragma unroll
    for (int nt = 0; nt < 4; ++nt) {
      int n     = n0 + wn + nt * 16 + r;
      int mbase = m0 + wm + mt * 16 + 8 * hh;   // 8 consecutive rows per lane
      float h[8];
      #pragma unroll
      for (int v = 0; v < 8; ++v) {
        float g = accg[mt][nt][v];
        float u = accu[mt][nt][v];
        h[v] = g * (1.0f / (1.0f + __expf(-g))) * u;   // silu(g)*u
      }
      u32x4 ov;
      ov[0] = pk_bf16(h[0], h[1]);
      ov[1] = pk_bf16(h[2], h[3]);
      ov[2] = pk_bf16(h[4], h[5]);
      ov[3] = pk_bf16(h[6], h[7]);
      *(u32x4*)&hsT[(size_t)n * TOKENS + mbase] = ov;   // global_store_b128
    }
  }
}

// ---------------------------------------------------------------------------
// Kernel 2: out[T,H] = hsT^T (bf16) @ Wd[e] (fp32->bf16), fp32 accumulate.
// ---------------------------------------------------------------------------
__global__ __launch_bounds__(256) void moe_down_kernel(
    const unsigned short* __restrict__ hsT, // [I,T] bf16
    const float* __restrict__ Wd,           // [E,I,H]
    const int*   __restrict__ gsz,          // [E]
    float* __restrict__ out)                // [T,H]
{
  __shared__ __align__(16) unsigned short As[BM * PADK];
  __shared__ __align__(16) unsigned short Bs[BN * PADK];

  const int m0 = blockIdx.x * BM;
  const int n0 = blockIdx.y * BN;

  const int e = expert_of_row(gsz, m0);
  const float* wd = Wd + (size_t)e * INTER * HIDDEN;

  const int tid  = threadIdx.x;
  const int lane = tid & 31;
  const int wid  = tid >> 5;
  const int r    = lane & 15;
  const int hh   = lane >> 4;
  const int wm   = (wid & 3) * 32;
  const int wn   = (wid >> 2) * 64;

  v8f acc[2][4];
  const v8f vzero = {0.f, 0.f, 0.f, 0.f, 0.f, 0.f, 0.f, 0.f};
  #pragma unroll
  for (int mt = 0; mt < 2; ++mt)
    #pragma unroll
    for (int nt = 0; nt < 4; ++nt) acc[mt][nt] = vzero;

  for (int kc = 0; kc < INTER; kc += BK) {
    // --- stage A: hsT rows k (64) x cols m (128) bf16 -> transpose to [m][k]
    #pragma unroll
    for (int p = 0; p < 8; ++p) {
      int idx = p * 256 + tid;
      int k  = idx >> 5;               // 32 quads of 4 bf16 per row
      int m4 = (idx & 31) << 2;
      u16x4 v = *(const u16x4*)&hsT[(size_t)(kc + k) * TOKENS + m0 + m4];
      #pragma unroll
      for (int j = 0; j < 4; ++j) As[(m4 + j) * PADK + k] = v[j];
    }
    // --- stage B: Wd 64x128 fp32 -> bf16 (packed cvt), transposed [n][k] ---
    #pragma unroll
    for (int p = 0; p < 8; ++p) {
      int idx = p * 256 + tid;
      int k  = idx >> 5;
      int c4 = (idx & 31) << 2;
      size_t go = (size_t)(kc + k) * HIDDEN + n0 + c4;
      f32x4 w = *(const f32x4*)(wd + go);
      if (kc + BK < INTER)
        __builtin_prefetch(wd + go + (size_t)BK * HIDDEN, 0, 0);
      unsigned w01 = pk_bf16(w[0], w[1]);
      unsigned w23 = pk_bf16(w[2], w[3]);
      Bs[(c4 + 0) * PADK + k] = (unsigned short)w01;
      Bs[(c4 + 1) * PADK + k] = (unsigned short)(w01 >> 16);
      Bs[(c4 + 2) * PADK + k] = (unsigned short)w23;
      Bs[(c4 + 3) * PADK + k] = (unsigned short)(w23 >> 16);
    }
    __syncthreads();

    #pragma unroll
    for (int ks = 0; ks < BK; ks += 32) {
      v16bf afrag[2];
      #pragma unroll
      for (int mt = 0; mt < 2; ++mt) {
        int base = (wm + mt * 16 + r) * PADK + ks + 8 * hh;
        afrag[mt] = make_frag(&As[base], &As[base + 16]);
      }
      #pragma unroll
      for (int nt = 0; nt < 4; ++nt) {
        int base = (wn + nt * 16 + r) * PADK + ks + 8 * hh;
        v16bf b = make_frag(&Bs[base], &Bs[base + 16]);
        #pragma unroll
        for (int mt = 0; mt < 2; ++mt) {
          acc[mt][nt] = __builtin_amdgcn_wmma_f32_16x16x32_bf16(
              false, afrag[mt], false, b, (short)0, acc[mt][nt], false, false);
        }
      }
    }
    __syncthreads();
  }

  // --- epilogue: fp32 out[T,H] ---
  #pragma unroll
  for (int mt = 0; mt < 2; ++mt) {
    #pragma unroll
    for (int nt = 0; nt < 4; ++nt) {
      int n     = n0 + wn + nt * 16 + r;
      int mbase = m0 + wm + mt * 16 + 8 * hh;
      #pragma unroll
      for (int v = 0; v < 8; ++v)
        out[(size_t)(mbase + v) * HIDDEN + n] = acc[mt][nt][v];
    }
  }
}

// ---------------------------------------------------------------------------
extern "C" void kernel_launch(void* const* d_in, const int* in_sizes, int n_in,
                              void* d_out, int out_size, void* d_ws, size_t ws_size,
                              hipStream_t stream) {
  (void)in_sizes; (void)n_in; (void)out_size; (void)ws_size;
  const float* x   = (const float*)d_in[0];
  const float* Wg  = (const float*)d_in[1];
  const float* Wu  = (const float*)d_in[2];
  const float* Wd  = (const float*)d_in[3];
  const int*   gsz = (const int*)d_in[4];
  float* outp = (float*)d_out;

  // workspace layout: xbf [T,H] bf16 (8 MB) | hsT [I,T] bf16 (16 MB)
  unsigned short* xbf = (unsigned short*)d_ws;
  unsigned short* hsT = xbf + (size_t)TOKENS * HIDDEN;

  dim3 block(256);
  dim3 grid0((TOKENS * HIDDEN) / (256 * 8));   // 2048
  dim3 grid1(TOKENS / BM, INTER / BN);         // 32 x 16
  dim3 grid2(TOKENS / BM, HIDDEN / BN);        // 32 x 8

  cvt_x_kernel<<<grid0, block, 0, stream>>>(x, xbf);
  moe_gateup_kernel<<<grid1, block, 0, stream>>>(xbf, Wg, Wu, gsz, hsT);
  moe_down_kernel<<<grid2, block, 0, stream>>>(hsT, Wd, gsz, outp);
}
